// MaskedIPAttnProcessor2_0_61993557950494
// MI455X (gfx1250) — compile-verified, matrix operand order checked by reading, not run on hardware
//
#include <hip/hip_runtime.h>

typedef unsigned short u16;
typedef unsigned int   u32;
typedef __attribute__((ext_vector_type(16))) __bf16 v16bf;
typedef __attribute__((ext_vector_type(8)))  float  v8f;

union Frag { v16bf v; uint4 q[2]; };

__device__ __forceinline__ u16 f2bf(float x) {
  u32 u = __float_as_uint(x);
  u32 r = u + 0x7FFFu + ((u >> 16) & 1u);   // round-to-nearest-even
  return (u16)(r >> 16);
}

__device__ __forceinline__ v8f wmma_bf16(const Frag& a, const Frag& b, v8f c) {
  return __builtin_amdgcn_wmma_f32_16x16x32_bf16(false, a.v, false, b.v,
                                                 (short)0, c, false, false);
}

// ---------------------------------------------------------------------------
// Generic tiled GEMM: C[M,N] = alpha * (A[M,K] @ B[K,N]) (+ bias)
// A fp32 row-major (lda=K), B fp32 row-major (ldb=N), C bf16 or fp32 (ldc=N).
// Rows >= Mload of A read as zero; rows < Mstore written (zero-padding rows).
// BM=128, BN=128, BK=32; 256 threads = 8 waves (4 along M, 2 along N),
// each wave computes a 32x64 block = 2x4 v_wmma_f32_16x16x32_bf16 tiles
// (8 WMMAs per k-step from 2 A-frags x 4 B-frags).
// ---------------------------------------------------------------------------
template<bool BF16OUT>
__global__ __launch_bounds__(256, 2)
void gemm_bf16_wmma(const float* __restrict__ A, long long strideAb,
                    const float* __restrict__ Bw,
                    void* __restrict__ Cv, long long strideCb,
                    int Mload, int Mstore, int K, int N,
                    float alpha, const float* __restrict__ bias)
{
  const int bz   = blockIdx.z;
  const float* Ab = A + (long long)bz * strideAb;
  const int tid  = threadIdx.x;
  const int wave = tid >> 5, lane = tid & 31;
  const int waveM = wave & 3, waveN = wave >> 2;
  const int r16 = lane & 15, hi = lane >> 4;
  const int ko  = hi * 8;               // halves offset of lo block
  const int m0  = blockIdx.y * 128;
  const int n0  = blockIdx.x * 128;

  __shared__ __align__(16) u16 sA[128 * 32];   // [row][k]
  __shared__ __align__(16) u16 sB[128 * 32];   // [n][k]

  const v8f vzero = {0.f,0.f,0.f,0.f,0.f,0.f,0.f,0.f};
  v8f c[2][4];
#pragma unroll
  for (int mt = 0; mt < 2; ++mt)
#pragma unroll
    for (int nt = 0; nt < 4; ++nt) c[mt][nt] = vzero;

  // A staging geometry: thread -> (row, 16-float chunk)
  const int arow = tid >> 1;
  const int ach  = (tid & 1) * 16;
  // B staging geometry: thread -> (k row, 16-column group)
  const int bk = tid >> 3;
  const int bn = (tid & 7) * 16;

  const int kTiles = K >> 5;
  for (int kt = 0; kt < kTiles; ++kt) {
    // ---- stage A tile: 128 rows x 32 k (fp32 -> bf16) ----
    {
      float4 f0, f1, f2, f3;
      if (m0 + arow < Mload) {
        const float* srcf = Ab + (long long)(m0 + arow) * K + kt * 32 + ach;
        const float4* src = (const float4*)srcf;
        f0 = src[0]; f1 = src[1]; f2 = src[2]; f3 = src[3];
        if (kt + 1 < kTiles) __builtin_prefetch(srcf + 32, 0, 1);  // next k-tile
      } else {
        f0 = make_float4(0,0,0,0); f1 = f0; f2 = f0; f3 = f0;
      }
      uint4 lo, hi4;
      lo.x = (u32)f2bf(f0.x) | ((u32)f2bf(f0.y) << 16);
      lo.y = (u32)f2bf(f0.z) | ((u32)f2bf(f0.w) << 16);
      lo.z = (u32)f2bf(f1.x) | ((u32)f2bf(f1.y) << 16);
      lo.w = (u32)f2bf(f1.z) | ((u32)f2bf(f1.w) << 16);
      hi4.x = (u32)f2bf(f2.x) | ((u32)f2bf(f2.y) << 16);
      hi4.y = (u32)f2bf(f2.z) | ((u32)f2bf(f2.w) << 16);
      hi4.z = (u32)f2bf(f3.x) | ((u32)f2bf(f3.y) << 16);
      hi4.w = (u32)f2bf(f3.z) | ((u32)f2bf(f3.w) << 16);
      uint4* dst = (uint4*)&sA[arow * 32 + ach];
      dst[0] = lo; dst[1] = hi4;
    }
    // ---- stage B tile: 32 k x 128 n, transposed to [n][k] ----
    {
      const float* srcf = Bw + (long long)(kt * 32 + bk) * N + n0 + bn;
      const float4* src = (const float4*)srcf;
      float4 b0 = src[0], b1 = src[1], b2 = src[2], b3 = src[3];
      if (kt + 1 < kTiles) __builtin_prefetch(srcf + (long long)32 * N, 0, 1);
      sB[(bn +  0) * 32 + bk] = f2bf(b0.x);
      sB[(bn +  1) * 32 + bk] = f2bf(b0.y);
      sB[(bn +  2) * 32 + bk] = f2bf(b0.z);
      sB[(bn +  3) * 32 + bk] = f2bf(b0.w);
      sB[(bn +  4) * 32 + bk] = f2bf(b1.x);
      sB[(bn +  5) * 32 + bk] = f2bf(b1.y);
      sB[(bn +  6) * 32 + bk] = f2bf(b1.z);
      sB[(bn +  7) * 32 + bk] = f2bf(b1.w);
      sB[(bn +  8) * 32 + bk] = f2bf(b2.x);
      sB[(bn +  9) * 32 + bk] = f2bf(b2.y);
      sB[(bn + 10) * 32 + bk] = f2bf(b2.z);
      sB[(bn + 11) * 32 + bk] = f2bf(b2.w);
      sB[(bn + 12) * 32 + bk] = f2bf(b3.x);
      sB[(bn + 13) * 32 + bk] = f2bf(b3.y);
      sB[(bn + 14) * 32 + bk] = f2bf(b3.z);
      sB[(bn + 15) * 32 + bk] = f2bf(b3.w);
    }
    __syncthreads();

    Frag af[2], bf[4];
#pragma unroll
    for (int mt = 0; mt < 2; ++mt) {
      const u16* p = &sA[(waveM * 32 + mt * 16 + r16) * 32 + ko];
      af[mt].q[0] = *(const uint4*)p;
      af[mt].q[1] = *(const uint4*)(p + 16);
    }
#pragma unroll
    for (int nt = 0; nt < 4; ++nt) {
      const u16* p = &sB[(waveN * 64 + nt * 16 + r16) * 32 + ko];
      bf[nt].q[0] = *(const uint4*)p;
      bf[nt].q[1] = *(const uint4*)(p + 16);
    }
#pragma unroll
    for (int mt = 0; mt < 2; ++mt)
#pragma unroll
      for (int nt = 0; nt < 4; ++nt)
        c[mt][nt] = wmma_bf16(af[mt], bf[nt], c[mt][nt]);
    __syncthreads();
  }

  // ---- store C ----
#pragma unroll
  for (int mt = 0; mt < 2; ++mt) {
#pragma unroll
    for (int j = 0; j < 8; ++j) {
      const int row = m0 + waveM * 32 + mt * 16 + j + 8 * hi;
      if (row < Mstore) {
#pragma unroll
        for (int nt = 0; nt < 4; ++nt) {
          const int col = n0 + waveN * 64 + nt * 16 + r16;
          float v = c[mt][nt][j] * alpha;
          if (BF16OUT) {
            ((u16*)Cv)[(long long)bz * strideCb + (long long)row * N + col] = f2bf(v);
          } else {
            if (bias) v += bias[col];
            ((float*)Cv)[(long long)bz * strideCb + (long long)row * N + col] = v;
          }
        }
      }
    }
  }
}

// ---------------------------------------------------------------------------
// Fused dual attention (text softmax + masked IP softmax, outputs summed).
// Grid: (Sq/64, heads=20, B=8). Block: 128 threads = 4 waves; each wave owns
// 16 query rows. All matmuls via v_wmma_f32_16x16x32_bf16.
// ---------------------------------------------------------------------------
__global__ __launch_bounds__(128, 2)
void attn_fused(const u16* __restrict__ qg,     // [B*4096,1280] bf16 (pre-scaled)
                const u16* __restrict__ kg,     // [B,80,1280] bf16 (rows 77..79 = 0)
                const u16* __restrict__ vg,     // [B,80,1280]
                const u16* __restrict__ kipg,   // [B,16,1280] (rows 12..15 = 0)
                const u16* __restrict__ vipg,   // [B,16,1280]
                const float* __restrict__ boxes,// [B,2,4]
                float* __restrict__ attn)       // [B*4096,1280] fp32
{
  const int b  = blockIdx.z;
  const int h  = blockIdx.y;
  const int qt = blockIdx.x;
  const int tid  = threadIdx.x;
  const int wave = tid >> 5, lane = tid & 31;
  const int r16 = lane & 15, hi = lane >> 4;
  const int ko  = hi * 8;

  __shared__ __align__(16) u16 sK[80 * 64];      // [kv][d]  (B-frag layout for QK^T)
  __shared__ __align__(16) u16 sKip[16 * 64];
  __shared__ __align__(16) u16 sVt[64 * 96];     // [d][kv]  (B-frag layout for P@V), kv padded 96
  __shared__ __align__(16) u16 sVipT[64 * 32];   // kv padded 32
  __shared__ __align__(16) u16 sP[4][16 * 96];   // per-wave probs (A-frag source)
  __shared__ __align__(16) u16 sPip[4][16 * 32];

  // ---- stage K / Kip (uint copies), Vt / VipT (transposed, zero-padded) ----
  for (int i = tid; i < 2560; i += 128) {        // 80*64 halves / 2
    const int kv = i >> 5, d2 = (i & 31) * 2;
    *(u32*)&sK[i * 2] = *(const u32*)&kg[((long long)(b * 80 + kv)) * 1280 + h * 64 + d2];
  }
  for (int i = tid; i < 512; i += 128) {         // 16*64 / 2
    const int kv = i >> 5, d2 = (i & 31) * 2;
    *(u32*)&sKip[i * 2] = *(const u32*)&kipg[((long long)(b * 16 + kv)) * 1280 + h * 64 + d2];
  }
  for (int i = tid; i < 64 * 96; i += 128) {
    const int d = i / 96, kv = i % 96;
    sVt[i] = (kv < 80) ? vg[((long long)(b * 80 + kv)) * 1280 + h * 64 + d] : (u16)0;
  }
  for (int i = tid; i < 64 * 32; i += 128) {
    const int d = i >> 5, kv = i & 31;
    sVipT[i] = (kv < 16) ? vipg[((long long)(b * 16 + kv)) * 1280 + h * 64 + d] : (u16)0;
  }
  __syncthreads();

  // ---- box mask params (P = sqrt(4096) = 64) ----
  const float* bx = boxes + b * 8;
  const int sx0 = (int)floorf(bx[0] * 64.f), sy0 = (int)floorf(bx[1] * 64.f);
  const int ex0 = (int)ceilf (bx[2] * 64.f), ey0 = (int)ceilf (bx[3] * 64.f);
  const int sx1 = (int)floorf(bx[4] * 64.f), sy1 = (int)floorf(bx[5] * 64.f);
  const int ex1 = (int)ceilf (bx[6] * 64.f), ey1 = (int)ceilf (bx[7] * 64.f);

  // ---- Q fragments straight from global (A-frag layout) ----
  const int qrow0 = qt * 64 + wave * 16;
  Frag qa[2];
  {
    const u16* qp = qg + ((long long)(b * 4096 + qrow0 + r16)) * 1280 + h * 64;
#pragma unroll
    for (int ks = 0; ks < 2; ++ks) {
      qa[ks].q[0] = *(const uint4*)(qp + ks * 32 + ko);
      qa[ks].q[1] = *(const uint4*)(qp + ks * 32 + ko + 16);
    }
  }

  const v8f vzero = {0.f,0.f,0.f,0.f,0.f,0.f,0.f,0.f};

  // ---- text logits: 16 x 80 (5 tiles), K = dh = 64 ----
  v8f t[5];
#pragma unroll
  for (int nt = 0; nt < 5; ++nt) t[nt] = vzero;
#pragma unroll
  for (int ks = 0; ks < 2; ++ks) {
#pragma unroll
    for (int nt = 0; nt < 5; ++nt) {
      Frag kb;
      const u16* p = &sK[(nt * 16 + r16) * 64 + ks * 32 + ko];
      kb.q[0] = *(const uint4*)p;
      kb.q[1] = *(const uint4*)(p + 16);
      t[nt] = wmma_bf16(qa[ks], kb, t[nt]);
    }
  }
#pragma unroll
  for (int nt = 0; nt < 5; ++nt) {
    const int col = nt * 16 + r16;
    if (col >= 77) {
#pragma unroll
      for (int j = 0; j < 8; ++j) t[nt][j] = -1e30f;
    }
  }

  // ---- IP logits: 16 x 16 (1 tile) + inline mask bias ----
  v8f tip = vzero;
#pragma unroll
  for (int ks = 0; ks < 2; ++ks) {
    Frag kb;
    const u16* p = &sKip[r16 * 64 + ks * 32 + ko];
    kb.q[0] = *(const uint4*)p;
    kb.q[1] = *(const uint4*)(p + 16);
    tip = wmma_bf16(qa[ks], kb, tip);
  }
#pragma unroll
  for (int j = 0; j < 8; ++j) {
    const int qpos = qrow0 + j + 8 * hi;
    const int y = qpos >> 6, x = qpos & 63;
    const float m0r = (x >= sx0 && x < ex0 && y >= sy0 && y < ey0) ? 1.f : 0.f;
    const float m1r = (x >= sx1 && x < ex1 && y >= sy1 && y < ey1) ? 1.f : 0.f;
    const int c = r16;
    if (c < 12) {
      float val;
      if (c < 4)      val = fmaxf(0.f, 1.f - m0r - m1r);   // dummy tokens
      else if (c < 8) val = m0r;                           // region 0 tokens
      else            val = m1r;                           // region 1 tokens
      tip[j] += (1.f - val) * (-10000.f);
    } else {
      tip[j] = -1e30f;                                     // padded columns
    }
  }

  // ---- softmax (text) -> sP (bf16, A-frag source layout, 96-wide padded) ----
#pragma unroll
  for (int j = 0; j < 8; ++j) {
    float m = t[0][j];
#pragma unroll
    for (int nt = 1; nt < 5; ++nt) m = fmaxf(m, t[nt][j]);
    m = fmaxf(m, __shfl_xor(m, 1, 32));
    m = fmaxf(m, __shfl_xor(m, 2, 32));
    m = fmaxf(m, __shfl_xor(m, 4, 32));
    m = fmaxf(m, __shfl_xor(m, 8, 32));
    float sum = 0.f;
#pragma unroll
    for (int nt = 0; nt < 5; ++nt) { float e = __expf(t[nt][j] - m); t[nt][j] = e; sum += e; }
    sum += __shfl_xor(sum, 1, 32);
    sum += __shfl_xor(sum, 2, 32);
    sum += __shfl_xor(sum, 4, 32);
    sum += __shfl_xor(sum, 8, 32);
    const float inv = __frcp_rn(sum);
    const int prow = (j + 8 * hi) * 96;
#pragma unroll
    for (int nt = 0; nt < 5; ++nt)
      sP[wave][prow + nt * 16 + r16] = f2bf(t[nt][j] * inv);
    sP[wave][prow + 80 + r16] = 0;                          // zero pad cols 80..95
  }

  // ---- softmax (IP) -> sPip (32-wide padded) ----
#pragma unroll
  for (int j = 0; j < 8; ++j) {
    float m = tip[j];
    m = fmaxf(m, __shfl_xor(m, 1, 32));
    m = fmaxf(m, __shfl_xor(m, 2, 32));
    m = fmaxf(m, __shfl_xor(m, 4, 32));
    m = fmaxf(m, __shfl_xor(m, 8, 32));
    float e = __expf(tip[j] - m);
    float sum = e;
    sum += __shfl_xor(sum, 1, 32);
    sum += __shfl_xor(sum, 2, 32);
    sum += __shfl_xor(sum, 4, 32);
    sum += __shfl_xor(sum, 8, 32);
    const int prow = (j + 8 * hi) * 32;
    sPip[wave][prow + r16] = f2bf(e * __frcp_rn(sum));
    sPip[wave][prow + 16 + r16] = 0;
  }

  // ---- out = probs @ V + probs_ip @ V_ip  (IP_SCALE = 1 folded in) ----
  v8f o[4];
#pragma unroll
  for (int nt = 0; nt < 4; ++nt) o[nt] = vzero;
#pragma unroll
  for (int ks = 0; ks < 3; ++ks) {                 // K = 96 (padded text kv)
    Frag pa;
    const u16* pp = &sP[wave][r16 * 96 + ks * 32 + ko];
    pa.q[0] = *(const uint4*)pp;
    pa.q[1] = *(const uint4*)(pp + 16);
#pragma unroll
    for (int nt = 0; nt < 4; ++nt) {
      Frag vb;
      const u16* vp = &sVt[(nt * 16 + r16) * 96 + ks * 32 + ko];
      vb.q[0] = *(const uint4*)vp;
      vb.q[1] = *(const uint4*)(vp + 16);
      o[nt] = wmma_bf16(pa, vb, o[nt]);
    }
  }
  {                                                // K = 32 (padded IP kv)
    Frag pa;
    const u16* pp = &sPip[wave][r16 * 32 + ko];
    pa.q[0] = *(const uint4*)pp;
    pa.q[1] = *(const uint4*)(pp + 16);
#pragma unroll
    for (int nt = 0; nt < 4; ++nt) {
      Frag vb;
      const u16* vp = &sVipT[(nt * 16 + r16) * 32 + ko];
      vb.q[0] = *(const uint4*)vp;
      vb.q[1] = *(const uint4*)(vp + 16);
      o[nt] = wmma_bf16(pa, vb, o[nt]);
    }
  }

  // ---- store merged-head fp32 output ----
#pragma unroll
  for (int j = 0; j < 8; ++j) {
    const long long row = (long long)(b * 4096 + qrow0 + j + 8 * hi);
#pragma unroll
    for (int nt = 0; nt < 4; ++nt) {
      const int col = h * 64 + nt * 16 + r16;
      attn[row * 1280 + col] = o[nt][j];
    }
  }
}

// ---------------------------------------------------------------------------
extern "C" void kernel_launch(void* const* d_in, const int* in_sizes, int n_in,
                              void* d_out, int out_size, void* d_ws, size_t ws_size,
                              hipStream_t stream) {
  const float* hidden = (const float*)d_in[0];
  const float* enc    = (const float*)d_in[1];
  const float* boxes  = (const float*)d_in[2];
  const float* Wq     = (const float*)d_in[3];
  const float* Wk     = (const float*)d_in[4];
  const float* Wv     = (const float*)d_in[5];
  const float* Wk_ip  = (const float*)d_in[6];
  const float* Wv_ip  = (const float*)d_in[7];
  const float* Wout   = (const float*)d_in[8];
  const float* bout   = (const float*)d_in[9];
  float* out = (float*)d_out;

  const int B = 8, SQ = 4096, H = 1280, C = 2048;
  const int M = B * SQ;                          // 32768

  char* ws = (char*)d_ws;
  size_t off = 0;
  auto alloc = [&](size_t bytes) -> void* {
    void* p = ws + off;
    off += (bytes + 255) & ~(size_t)255;
    return p;
  };
  u16*  qb   = (u16*)alloc((size_t)M * H * 2);         // scaled Q, bf16
  u16*  kb   = (u16*)alloc((size_t)B * 80 * H * 2);
  u16*  vb   = (u16*)alloc((size_t)B * 80 * H * 2);
  u16*  kipb = (u16*)alloc((size_t)B * 16 * H * 2);
  u16*  vipb = (u16*)alloc((size_t)B * 16 * H * 2);
  float* attn = (float*)alloc((size_t)M * H * 4);

  dim3 blk(256);
  const float sc = 0.125f;                             // dh^-0.5, dh = 64

  // Q = (hidden @ Wq) * sc  -> bf16
  gemm_bf16_wmma<true><<<dim3(H / 128, M / 128, 1), blk, 0, stream>>>(
      hidden, 0, Wq, qb, 0, M, M, H, H, sc, nullptr);
  // K / V from text tokens (77 rows, zero-padded to 80)
  gemm_bf16_wmma<true><<<dim3(H / 128, 1, B), blk, 0, stream>>>(
      enc, (long long)89 * C, Wk, kb, (long long)80 * H, 77, 80, C, H, 1.0f, nullptr);
  gemm_bf16_wmma<true><<<dim3(H / 128, 1, B), blk, 0, stream>>>(
      enc, (long long)89 * C, Wv, vb, (long long)80 * H, 77, 80, C, H, 1.0f, nullptr);
  // K_ip / V_ip from IP tokens (12 rows, zero-padded to 16)
  gemm_bf16_wmma<true><<<dim3(H / 128, 1, B), blk, 0, stream>>>(
      enc + 77 * C, (long long)89 * C, Wk_ip, kipb, (long long)16 * H, 12, 16, C, H, 1.0f, nullptr);
  gemm_bf16_wmma<true><<<dim3(H / 128, 1, B), blk, 0, stream>>>(
      enc + 77 * C, (long long)89 * C, Wv_ip, vipb, (long long)16 * H, 12, 16, C, H, 1.0f, nullptr);

  // fused text + IP attention
  attn_fused<<<dim3(SQ / 64, 20, B), dim3(128), 0, stream>>>(
      qb, kb, vb, kipb, vipb, boxes, attn);

  // out = attn @ Wout + bout  (fp32)
  gemm_bf16_wmma<false><<<dim3(H / 128, M / 128, 1), blk, 0, stream>>>(
      attn, 0, Wout, out, 0, M, M, H, H, 1.0f, bout);
}
